// MultiLabelEmbedding_64106681860445
// MI455X (gfx1250) — matrix operand order repaired and checked by reading, not compiled, not run on hardware
//
#include <hip/hip_runtime.h>

// GEMM: Out[M,N] = X[M,K] * W[K,N], fp32 in/out, M=32768 K=512 N=1024.
// Split-bf16 (hi+lo) -> 3x v_wmma_f32_16x16x32_bf16 with fp32 accumulation.
// W is pre-split once into unpadded LDS-image tiles in d_ws; the main loop pulls
// each 16KB W tile with one TENSOR_LOAD_TO_LDS (TDM) and double-buffers LDS.

typedef __attribute__((ext_vector_type(16))) __bf16 v16bf;
typedef __attribute__((ext_vector_type(8)))  float  v8f;
typedef __attribute__((ext_vector_type(4)))  unsigned int uv4;
typedef __attribute__((ext_vector_type(8)))  int iv8;
typedef __attribute__((ext_vector_type(4)))  int iv4;

#define M_DIM 32768
#define N_DIM 1024
#define K_DIM 512
#define BM 128
#define BN 128
#define BK 32
#define KT_STEPS (K_DIM / BK)        // 16
#define LDA 20                        // uints per LDS row: 16 k-pairs + 4 pad
#define LDB 20
#define HALF_TILE (BM * LDA)          // 2560 uints (hi region incl. pad)
#define TILE_U (2 * HALF_TILE)        // 5120 uints per pipeline buffer (hi+lo)
#define WTILE_DW 4096                 // dwords per packed W tile (hi 2048 + lo 2048)
#define WS_NEEDED ((size_t)(N_DIM / BN) * KT_STEPS * WTILE_DW * 4)  // 2 MB

union FragU { uint4 q[2]; v16bf v; };

// Split two consecutive-K fp32 values into packed (hi,hi) and (lo,lo) bf16 pairs.
// K-even in bits [15:0], K-odd in [31:16] (matches WMMA 16-bit VGPR layout).
__device__ __forceinline__ void split2(float a, float b, unsigned& uhi, unsigned& ulo) {
    __bf16 ah = (__bf16)a;
    __bf16 bh = (__bf16)b;
    float ar = a - (float)ah;
    float br = b - (float)bh;
    __bf16 al = (__bf16)ar;
    __bf16 bl = (__bf16)br;
    unsigned short ahu = __builtin_bit_cast(unsigned short, ah);
    unsigned short bhu = __builtin_bit_cast(unsigned short, bh);
    unsigned short alu = __builtin_bit_cast(unsigned short, al);
    unsigned short blu = __builtin_bit_cast(unsigned short, bl);
    uhi = ((unsigned)bhu << 16) | (unsigned)ahu;
    ulo = ((unsigned)blu << 16) | (unsigned)alu;
}

// ---------------- prep kernel: W (512x1024 f32) -> packed bf16 hi/lo tiles ----------------
// Tile (bn, kt): 128 n-cols x 16 k-pairs, [n][pair] uint, hi 2048 dw then lo 2048 dw,
// contiguous 16KB — a byte-exact image of the (unpadded) LDS B tile.
__global__ __launch_bounds__(256)
void MLE_prep_w(const float* __restrict__ W, unsigned* __restrict__ Wp) {
    const int b  = blockIdx.x;      // 0..127
    const int bn = b >> 4;          // 0..7
    const int kt = b & 15;          // 0..15
    const int t  = threadIdx.x;
    const int n  = t >> 1;          // 0..127
    const int ph = (t & 1) * 8;     // pair half: 0 or 8
    unsigned h[8], l[8];
    #pragma unroll
    for (int j = 0; j < 8; ++j) {
        int p = ph + j;
        int k = kt * BK + 2 * p;
        float w0 = W[(size_t)k       * N_DIM + bn * BN + n];
        float w1 = W[(size_t)(k + 1) * N_DIM + bn * BN + n];
        split2(w0, w1, h[j], l[j]);
    }
    unsigned* base = Wp + (size_t)(bn * KT_STEPS + kt) * WTILE_DW;
    *(uint4*)(base + n * 16 + ph)            = uint4{h[0], h[1], h[2], h[3]};
    *(uint4*)(base + n * 16 + ph + 4)        = uint4{h[4], h[5], h[6], h[7]};
    *(uint4*)(base + 2048 + n * 16 + ph)     = uint4{l[0], l[1], l[2], l[3]};
    *(uint4*)(base + 2048 + n * 16 + ph + 4) = uint4{l[4], l[5], l[6], l[7]};
}

// ---------------- TDM: one descriptor copies a 16KB W tile into LDS with padding ----------
// pad_enable: after every 16 dwords insert 4 dwords -> LDS row stride 20 uints, and the
// hi region lands on exactly HALF_TILE uints so lo continues at sB + HALF_TILE.
__device__ __forceinline__ void tdm_load_wtile(unsigned lds_byte_off, const unsigned* gsrc) {
    unsigned long long ga = (unsigned long long)(uintptr_t)gsrc;
    uv4 g0 = { 1u,                                        // count=1 (valid descriptor)
               lds_byte_off,                              // lds_addr
               (unsigned)ga,                              // global_addr[31:0]
               (unsigned)((ga >> 32) & 0x1FFFFFFu)        // global_addr[56:32]
                   | (2u << 30) };                        // type=2 ("image")
    iv8 g1 = { (int)((2u << 16)      // data_size = 4 bytes
                   | (1u << 20)      // pad_enable
                   | (3u << 22)      // pad_interval: 16 dwords
                   | (3u << 25)),    // pad_amount:   4 dwords
               (int)(4096u << 16),   // tensor_dim0 = 4096 (bits [79:48], low 16 here)
               (int)(1u << 16),      // tensor_dim1 = 1
               (int)(4096u << 16),   // tile_dim0 = 4096 (bits [127:112])
               0,                    // tile_dim1 = 0 (unused), tile_dim2 = 0
               4096,                 // tensor_dim0_stride low 32
               0, 0 };
    iv4 gz4 = { 0, 0, 0, 0 };
    iv8 gz8 = { 0, 0, 0, 0, 0, 0, 0, 0 };
    __builtin_amdgcn_tensor_load_to_lds(g0, g1, gz4, gz4, gz8, 0);
}

// ---------------- per-wave fragment loads + 24 WMMAs -------------------------------------
__device__ __forceinline__ void compute_tile(const unsigned* __restrict__ sBhi,
                                             const unsigned* __restrict__ sBlo,
                                             const unsigned* __restrict__ sAhi,
                                             const unsigned* __restrict__ sAlo,
                                             int wm, int wn, int lane16, int half,
                                             v8f (&acc)[2][4]) {
    // B fragments: lanes 0-15 hold K=0..15 (pairs 0..7), lanes 16-31 K=16..31
    FragU bhi[4], blo[4];
    const int pB = half * 8;
    #pragma unroll
    for (int j = 0; j < 4; ++j) {
        const int n = wn + j * 16 + lane16;
        bhi[j].q[0] = *(const uint4*)&sBhi[n * LDB + pB];
        bhi[j].q[1] = *(const uint4*)&sBhi[n * LDB + pB + 4];
        blo[j].q[0] = *(const uint4*)&sBlo[n * LDB + pB];
        blo[j].q[1] = *(const uint4*)&sBlo[n * LDB + pB + 4];
    }
    // A fragments: lanes 0-15 pairs 0..7; lanes 16-31 pairs 8..15
    const int pA = half * 8;
    #pragma unroll
    for (int i = 0; i < 2; ++i) {
        const int m = wm + i * 16 + lane16;
        FragU ahi, alo;
        ahi.q[0] = *(const uint4*)&sAhi[m * LDA + pA];
        ahi.q[1] = *(const uint4*)&sAhi[m * LDA + pA + 4];
        alo.q[0] = *(const uint4*)&sAlo[m * LDA + pA];
        alo.q[1] = *(const uint4*)&sAlo[m * LDA + pA + 4];
        #pragma unroll
        for (int j = 0; j < 4; ++j) {
            acc[i][j] = __builtin_amdgcn_wmma_f32_16x16x32_bf16(
                false, ahi.v, false, bhi[j].v, (short)0, acc[i][j], false, false);
            acc[i][j] = __builtin_amdgcn_wmma_f32_16x16x32_bf16(
                false, ahi.v, false, blo[j].v, (short)0, acc[i][j], false, false);
            acc[i][j] = __builtin_amdgcn_wmma_f32_16x16x32_bf16(
                false, alo.v, false, bhi[j].v, (short)0, acc[i][j], false, false);
        }
    }
}

// ---------------- main kernel: TDM-fed, double-buffered, software-pipelined ---------------
__global__ __launch_bounds__(256, 2)
void MLE_gemm_tdm(const float* __restrict__ X,
                  const unsigned* __restrict__ Wp,
                  float* __restrict__ Out) {
    __shared__ unsigned sA[2][TILE_U];  // [buf][hi 2560 | lo 2560]
    __shared__ unsigned sB[2][TILE_U];

    const int t    = threadIdx.x;
    const int bn0  = blockIdx.x * BN;
    const int bm0  = blockIdx.y * BM;

    const int waveId = t >> 5;
    const int lane   = t & 31;
    const int lane16 = lane & 15;
    const int half   = lane >> 4;
    const int wm     = (waveId & 3) * 32;
    const int wn     = (waveId >> 2) * 64;

    // A staging coords: 2 threads per row, each converts 16 consecutive K (8 pairs)
    const int am  = t >> 1;
    const int akp = (t & 1) * 8;
    const float* xrow = X + (size_t)(bm0 + am) * K_DIM + (t & 1) * 16;

    const unsigned* wTiles = Wp + (size_t)blockIdx.x * KT_STEPS * WTILE_DW;

    const v8f vzero = {0.f, 0.f, 0.f, 0.f, 0.f, 0.f, 0.f, 0.f};
    v8f acc[2][4];
    #pragma unroll
    for (int i = 0; i < 2; ++i)
        #pragma unroll
        for (int j = 0; j < 4; ++j)
            acc[i][j] = vzero;

    // ---- prologue: stage tile 0 ----
    {
        const float4* xp = (const float4*)(xrow);
        float4 f0 = xp[0], f1 = xp[1], f2 = xp[2], f3 = xp[3];
        unsigned h[8], l[8];
        split2(f0.x, f0.y, h[0], l[0]); split2(f0.z, f0.w, h[1], l[1]);
        split2(f1.x, f1.y, h[2], l[2]); split2(f1.z, f1.w, h[3], l[3]);
        split2(f2.x, f2.y, h[4], l[4]); split2(f2.z, f2.w, h[5], l[5]);
        split2(f3.x, f3.y, h[6], l[6]); split2(f3.z, f3.w, h[7], l[7]);
        *(uint4*)&sA[0][am * LDA + akp]                 = uint4{h[0], h[1], h[2], h[3]};
        *(uint4*)&sA[0][am * LDA + akp + 4]             = uint4{h[4], h[5], h[6], h[7]};
        *(uint4*)&sA[0][HALF_TILE + am * LDA + akp]     = uint4{l[0], l[1], l[2], l[3]};
        *(uint4*)&sA[0][HALF_TILE + am * LDA + akp + 4] = uint4{l[4], l[5], l[6], l[7]};
        if (waveId == 0)
            tdm_load_wtile((unsigned)(uintptr_t)&sB[0][0], wTiles);
    }

    for (int kt = 0; kt < KT_STEPS; ++kt) {
        const int cur = kt & 1;
        const int nxt = cur ^ 1;
        if (waveId == 0)
            __builtin_amdgcn_s_wait_tensorcnt(0);   // TDM for 'cur' done
        __syncthreads();                             // A stores for 'cur' visible too

        const bool hasNext = (kt + 1) < KT_STEPS;
        float4 f0, f1, f2, f3;
        if (hasNext) {
            const float4* xp = (const float4*)(xrow + (size_t)(kt + 1) * BK);
            f0 = xp[0]; f1 = xp[1]; f2 = xp[2]; f3 = xp[3];   // issue early
            if (waveId == 0)
                tdm_load_wtile((unsigned)(uintptr_t)&sB[nxt][0],
                               wTiles + (size_t)(kt + 1) * WTILE_DW);
        }

        compute_tile(&sB[cur][0], &sB[cur][HALF_TILE],
                     &sA[cur][0], &sA[cur][HALF_TILE],
                     wm, wn, lane16, half, acc);

        if (hasNext) {   // split + store co-execute behind the XDL pipe
            unsigned h[8], l[8];
            split2(f0.x, f0.y, h[0], l[0]); split2(f0.z, f0.w, h[1], l[1]);
            split2(f1.x, f1.y, h[2], l[2]); split2(f1.z, f1.w, h[3], l[3]);
            split2(f2.x, f2.y, h[4], l[4]); split2(f2.z, f2.w, h[5], l[5]);
            split2(f3.x, f3.y, h[6], l[6]); split2(f3.z, f3.w, h[7], l[7]);
            *(uint4*)&sA[nxt][am * LDA + akp]                 = uint4{h[0], h[1], h[2], h[3]};
            *(uint4*)&sA[nxt][am * LDA + akp + 4]             = uint4{h[4], h[5], h[6], h[7]};
            *(uint4*)&sA[nxt][HALF_TILE + am * LDA + akp]     = uint4{l[0], l[1], l[2], l[3]};
            *(uint4*)&sA[nxt][HALF_TILE + am * LDA + akp + 4] = uint4{l[4], l[5], l[6], l[7]};
        }
    }

    // ---- epilogue: C layout -> VGPR r: M = r (lanes 0-15) / 8+r (lanes 16-31) ----
    #pragma unroll
    for (int i = 0; i < 2; ++i) {
        #pragma unroll
        for (int j = 0; j < 4; ++j) {
            const int col = bn0 + wn + j * 16 + lane16;
            #pragma unroll
            for (int r = 0; r < 8; ++r) {
                const int row = bm0 + wm + i * 16 + half * 8 + r;
                Out[(size_t)row * N_DIM + col] = acc[i][j][r];
            }
        }
    }
}

// ---------------- fallback (no workspace): round-1 all-in-one kernel ----------------------
__global__ __launch_bounds__(256, 2)
void MLE_gemm_fallback(const float* __restrict__ X,
                       const float* __restrict__ W,
                       float* __restrict__ Out) {
    __shared__ unsigned sAhi[BM * LDA], sAlo[BM * LDA], sBhi[BN * LDB], sBlo[BN * LDB];

    const int t    = threadIdx.x;
    const int bn0  = blockIdx.x * BN;
    const int bm0  = blockIdx.y * BM;
    const int waveId = t >> 5;
    const int lane   = t & 31;
    const int lane16 = lane & 15;
    const int half   = lane >> 4;
    const int wm     = (waveId & 3) * 32;
    const int wn     = (waveId >> 2) * 64;
    const int am  = t >> 1;
    const int akp = (t & 1) * 8;
    const float* xrow = X + (size_t)(bm0 + am) * K_DIM + (t & 1) * 16;
    const int bnn = t & 127;
    const int bkp = (t >> 7) * 8;

    const v8f vzero = {0.f, 0.f, 0.f, 0.f, 0.f, 0.f, 0.f, 0.f};
    v8f acc[2][4];
    #pragma unroll
    for (int i = 0; i < 2; ++i)
        #pragma unroll
        for (int j = 0; j < 4; ++j)
            acc[i][j] = vzero;

    for (int k0 = 0; k0 < K_DIM; k0 += BK) {
        unsigned h[8], l[8];
        {
            const float4* xp = (const float4*)(xrow + k0);
            float4 f0 = xp[0], f1 = xp[1], f2 = xp[2], f3 = xp[3];
            if (k0 + BK < K_DIM) __builtin_prefetch(xrow + k0 + BK, 0, 0);
            split2(f0.x, f0.y, h[0], l[0]); split2(f0.z, f0.w, h[1], l[1]);
            split2(f1.x, f1.y, h[2], l[2]); split2(f1.z, f1.w, h[3], l[3]);
            split2(f2.x, f2.y, h[4], l[4]); split2(f2.z, f2.w, h[5], l[5]);
            split2(f3.x, f3.y, h[6], l[6]); split2(f3.z, f3.w, h[7], l[7]);
            *(uint4*)&sAhi[am * LDA + akp]     = uint4{h[0], h[1], h[2], h[3]};
            *(uint4*)&sAhi[am * LDA + akp + 4] = uint4{h[4], h[5], h[6], h[7]};
            *(uint4*)&sAlo[am * LDA + akp]     = uint4{l[0], l[1], l[2], l[3]};
            *(uint4*)&sAlo[am * LDA + akp + 4] = uint4{l[4], l[5], l[6], l[7]};
        }
        {
            #pragma unroll
            for (int j = 0; j < 8; ++j) {
                int k = k0 + bkp * 2 + 2 * j;
                float w0 = W[(size_t)k       * N_DIM + bn0 + bnn];
                float w1 = W[(size_t)(k + 1) * N_DIM + bn0 + bnn];
                split2(w0, w1, h[j], l[j]);
            }
            *(uint4*)&sBhi[bnn * LDB + bkp]     = uint4{h[0], h[1], h[2], h[3]};
            *(uint4*)&sBhi[bnn * LDB + bkp + 4] = uint4{h[4], h[5], h[6], h[7]};
            *(uint4*)&sBlo[bnn * LDB + bkp]     = uint4{l[0], l[1], l[2], l[3]};
            *(uint4*)&sBlo[bnn * LDB + bkp + 4] = uint4{l[4], l[5], l[6], l[7]};
        }
        __syncthreads();
        compute_tile(sBhi, sBlo, sAhi, sAlo, wm, wn, lane16, half, acc);
        __syncthreads();
    }

    #pragma unroll
    for (int i = 0; i < 2; ++i) {
        #pragma unroll
        for (int j = 0; j < 4; ++j) {
            const int col = bn0 + wn + j * 16 + lane16;
            #pragma unroll
            for (int r = 0; r < 8; ++r) {
                const int row = bm0 + wm + i * 16 + half * 8 + r;
                Out[(size_t)row * N_DIM + col] = acc[i][j][r];
            }
        }
    }
}

extern "C" void kernel_launch(void* const* d_in, const int* in_sizes, int n_in,
                              void* d_out, int out_size, void* d_ws, size_t ws_size,
                              hipStream_t stream) {
    (void)in_sizes; (void)n_in; (void)out_size;
    const float* x = (const float*)d_in[0];   // (32768, 512) fp32
    const float* w = (const float*)d_in[1];   // (512, 1024) fp32
    float* out = (float*)d_out;               // (32768, 1024) fp32

    dim3 grid(N_DIM / BN, M_DIM / BM);        // 8 x 256
    dim3 block(256);                          // 8 waves (wave32)

    if (d_ws != nullptr && ws_size >= WS_NEEDED) {
        MLE_prep_w<<<dim3((N_DIM / BN) * KT_STEPS), block, 0, stream>>>(w, (unsigned*)d_ws);
        MLE_gemm_tdm<<<grid, block, 0, stream>>>(x, (const unsigned*)d_ws, out);
    } else {
        MLE_gemm_fallback<<<grid, block, 0, stream>>>(x, w, out);
    }
}
